// MambaBlock_63342177681522
// MI455X (gfx1250) — matrix operand order repaired and checked by reading, not compiled, not run on hardware
//
#include <hip/hip_runtime.h>
#include <cstddef>
#include <cstdint>

#define D_MODEL 256
#define D_INNER 512
#define D_STATE 16
#define BATCH   2
#define SEQ     4096
#define BL      (BATCH * SEQ)          // 8192 tokens in scan order
#define NBIG    576                    // 48 (dbl) + 512 (dt_pre) + 16 pad
#define CHUNK   64
#define NCHUNK  (SEQ / CHUNK)          // 64

typedef __attribute__((ext_vector_type(16))) __bf16 v16bf;
typedef __attribute__((ext_vector_type(8)))  float  v8f;

__device__ __forceinline__ unsigned short f2bf(float f) {
  unsigned u = __float_as_uint(f);
  unsigned r = u + 0x7FFFu + ((u >> 16) & 1u);   // round-to-nearest-even
  return (unsigned short)(r >> 16);
}

union Frag { uint4 q[2]; v16bf v; };

// async global->LDS copy of 16 bytes (per lane), tracked by ASYNCcnt
__device__ __forceinline__ void async_b128(unsigned ldsOff, const float* gp) {
  asm volatile("global_load_async_to_lds_b128 %0, %1, off"
               :: "v"(ldsOff), "v"(gp) : "memory");
}
__device__ __forceinline__ void wait_async0() {
  asm volatile("s_wait_asynccnt 0x0" ::: "memory");
}

// ---------------------------------------------------------------- RMSNorm -> bf16 (scan order, optional flip)
__global__ __launch_bounds__(256)
void k_rmsnorm(const float* __restrict__ x, const float* __restrict__ w,
               unsigned short* __restrict__ xn, int flip) {
  __shared__ float red[256];
  int t = blockIdx.x;                       // scan-order token
  int b = t / SEQ, tl = t - b * SEQ;
  int src = flip ? (b * SEQ + (SEQ - 1 - tl)) : t;
  float v = x[(size_t)src * D_MODEL + threadIdx.x];
  red[threadIdx.x] = v * v;
  __syncthreads();
  for (int s = 128; s > 0; s >>= 1) {
    if ((int)threadIdx.x < s) red[threadIdx.x] += red[threadIdx.x + s];
    __syncthreads();
  }
  float r = rsqrtf(red[0] * (1.0f / D_MODEL) + 1e-5f);
  xn[(size_t)t * D_MODEL + threadIdx.x] = f2bf(v * r * w[threadIdx.x]);
}

// ---------------------------------------------------------------- flat f32 -> bf16
__global__ __launch_bounds__(256)
void k_f2bf(const float* __restrict__ in, unsigned short* __restrict__ out, int n) {
  int i = blockIdx.x * 256 + threadIdx.x;
  if (i < n) out[i] = f2bf(in[i]);
}

// ---------------------------------------------------------------- build padded (576,512) weight:
// rows 0..47 = x_proj; rows 48..559 = dt_w @ x_proj[:16]; rows 560..575 = 0
__global__ __launch_bounds__(256)
void k_wbig(const float* __restrict__ xp, const float* __restrict__ dtw,
            unsigned short* __restrict__ wbig) {
  int idx = blockIdx.x * 256 + threadIdx.x;
  if (idx >= NBIG * D_INNER) return;
  int k = idx & (D_INNER - 1);
  int row = idx >> 9;
  float v = 0.0f;
  if (row < 48) {
    v = xp[row * D_INNER + k];
  } else if (row < 560) {
    int d = row - 48;
    float s = 0.0f;
    #pragma unroll
    for (int r = 0; r < 16; ++r) s += dtw[d * 16 + r] * xp[r * D_INNER + k];
    v = s;
  }
  wbig[idx] = f2bf(v);
}

// ---------------------------------------------------------------- bf16 WMMA GEMM: C[M,N] = A[M,K] * W[N,K]^T
// block = 256 threads = 8 waves arranged (8/nWaves) x nWaves; each wave owns a
// 32x16 tile (2 accumulators sharing one B fragment -> 2 WMMA per fragment set)
__global__ __launch_bounds__(256)
void k_gemm(const unsigned short* __restrict__ A, int lda,
            const unsigned short* __restrict__ W, int ldw,
            float* __restrict__ C, int ldc, int K, int nWaves) {
  int lane = threadIdx.x & 31;
  int wave = threadIdx.x >> 5;
  int mWaves = 8 / nWaves;
  int wm = wave / nWaves, wn = wave - wm * nWaves;
  int row0 = (blockIdx.x * mWaves + wm) * 32;
  int col0 = (blockIdx.y * nWaves + wn) * 16;
  int half = lane >> 4;
  int mr   = lane & 15;     // A row within tile; also B column / C column

  v8f acc0 = {}, acc1 = {};
  const unsigned short* arow0 = A + (size_t)(row0 + mr) * lda;
  const unsigned short* arow1 = arow0 + (size_t)16 * lda;
  const unsigned short* wrow  = W + (size_t)(col0 + mr) * ldw;

  for (int k0 = 0; k0 < K; k0 += 32) {
    Frag fa0, fa1, fb;
    // A 16x32 bf16: lanes<16 hold K {0..7,16..23}, lanes>=16 hold K {8..15,24..31}
    fa0.q[0] = *(const uint4*)(arow0 + k0 + half * 8);
    fa0.q[1] = *(const uint4*)(arow0 + k0 + 16 + half * 8);
    fa1.q[0] = *(const uint4*)(arow1 + k0 + half * 8);
    fa1.q[1] = *(const uint4*)(arow1 + k0 + 16 + half * 8);
    // B 32x16 bf16: lane n holds column n; lanes<16 K {0..7,16..23}, lanes>=16 K {8..15,24..31}
    fb.q[0] = *(const uint4*)(wrow + k0 + half * 8);
    fb.q[1] = *(const uint4*)(wrow + k0 + 16 + half * 8);
    acc0 = __builtin_amdgcn_wmma_f32_16x16x32_bf16(
        false, fa0.v, false, fb.v, (short)0, acc0, false, false);
    acc1 = __builtin_amdgcn_wmma_f32_16x16x32_bf16(
        false, fa1.v, false, fb.v, (short)0, acc1, false, false);
  }
  // C/D 16x16 f32: VGPR r -> row r + 8*half, column = lane&15
  float* crow0 = C + (size_t)(row0 + 8 * half) * ldc + col0 + mr;
  float* crow1 = crow0 + (size_t)16 * ldc;
  #pragma unroll
  for (int r = 0; r < 8; ++r) {
    crow0[(size_t)r * ldc] = acc0[r];
    crow1[(size_t)r * ldc] = acc1[r];
  }
}

// ---------------------------------------------------------------- causal depthwise conv4 + SiLU
__global__ __launch_bounds__(256)
void k_conv(const float* __restrict__ xiz, const float* __restrict__ cw,
            const float* __restrict__ cb, float* __restrict__ xc,
            unsigned short* __restrict__ xcb) {
  int idx = blockIdx.x * 256 + threadIdx.x;    // BL * 512
  int d = idx & (D_INNER - 1);
  int t = idx >> 9;
  int b = t / SEQ, tl = t - b * SEQ;
  float acc = cb[d];
  #pragma unroll
  for (int j = 0; j < 4; ++j) {
    int tt = tl - 3 + j;
    if (tt >= 0) acc += cw[d * 4 + j] * xiz[(size_t)(b * SEQ + tt) * 1024 + d];
  }
  float v = acc / (1.0f + __expf(-acc));       // SiLU
  xc[idx] = v;
  xcb[idx] = f2bf(v);
}

// ---------------------------------------------------------------- dt = softplus(dt_pre + dt_b)
__global__ __launch_bounds__(256)
void k_dt(const float* __restrict__ dbl, const float* __restrict__ dtb,
          float* __restrict__ dt) {
  int idx = blockIdx.x * 256 + threadIdx.x;    // BL * 512
  int d = idx & (D_INNER - 1);
  int t = idx >> 9;
  float p = dbl[(size_t)t * NBIG + 48 + d] + dtb[d];
  dt[idx] = (p > 20.0f) ? p : log1pf(__expf(p));
}

// ---------------------------------------------------------------- async staging of one scan chunk into LDS
// sdt/sxc: [CHUNK][16] (this block's 16 channels); sbc: [CHUNK][32] (B|C)
__device__ __forceinline__ void stage_chunk(const float* __restrict__ xc,
                                            const float* __restrict__ dt,
                                            const float* __restrict__ dbl,
                                            size_t t0, int d0, int tid,
                                            float (*sdt)[16], float (*sxc)[16],
                                            float (*sbc)[32]) {
  int t = tid >> 2, g = (tid & 3) * 4;                  // 256 thr -> 64 rows x 4 groups
  async_b128((unsigned)(uintptr_t)&sdt[t][g], dt + (t0 + t) * D_INNER + d0 + g);
  async_b128((unsigned)(uintptr_t)&sxc[t][g], xc + (t0 + t) * D_INNER + d0 + g);
  #pragma unroll
  for (int rep = 0; rep < 2; ++rep) {                   // 512 xfers of B|C
    int idx = rep * 256 + tid;
    int tt = idx >> 3, gg = (idx & 7) * 4;
    async_b128((unsigned)(uintptr_t)&sbc[tt][gg], dbl + (t0 + tt) * NBIG + 16 + gg);
  }
  wait_async0();
  __syncthreads();
}

// ---------------------------------------------------------------- scan pass 1: per-chunk decay + local state
__global__ __launch_bounds__(256)
void k_scan1(const float* __restrict__ xc, const float* __restrict__ dt,
             const float* __restrict__ dbl, const float* __restrict__ A_log,
             float* __restrict__ cA, float* __restrict__ cH) {
  __shared__ __align__(16) float sdt[CHUNK][16];
  __shared__ __align__(16) float sxc[CHUNK][16];
  __shared__ __align__(16) float sbc[CHUNK][32];
  int tid = threadIdx.x;
  int n  = tid & 15;
  int dl = tid >> 4;
  int b  = blockIdx.y >> 5;
  int d0 = (blockIdx.y & 31) * 16;
  int d  = d0 + dl;
  int c  = blockIdx.x;
  size_t t0 = (size_t)b * SEQ + (size_t)c * CHUNK;
  stage_chunk(xc, dt, dbl, t0, d0, tid, sdt, sxc, sbc);

  float An = -__expf(A_log[d * 16 + n]);
  float h = 0.0f, s = 0.0f;
  for (int i = 0; i < CHUNK; ++i) {
    float dtv = sdt[i][dl];
    float u   = sxc[i][dl];
    float Bv  = sbc[i][n];
    float xA  = dtv * An;
    s += xA;
    h = __expf(xA) * h + dtv * u * Bv;
  }
  size_t o = ((((size_t)b * NCHUNK + c) * D_INNER) + d) * D_STATE + n;
  cA[o] = __expf(s);   // s <= 0, safe
  cH[o] = h;
}

// ---------------------------------------------------------------- scan pass 2: stitch chunk states (depth 64)
__global__ __launch_bounds__(256)
void k_scan2(const float* __restrict__ cA, const float* __restrict__ cH,
             float* __restrict__ Hst) {
  int n  = threadIdx.x & 15;
  int dl = threadIdx.x >> 4;
  int b  = blockIdx.x >> 5;
  int d  = ((blockIdx.x & 31) * 16) + dl;
  float H = 0.0f;
  for (int c = 0; c < NCHUNK; ++c) {
    size_t o = ((((size_t)b * NCHUNK + c) * D_INNER) + d) * D_STATE + n;
    if (c + 1 < NCHUNK) {
      __builtin_prefetch(cA + o + (size_t)D_INNER * D_STATE, 0, 1);
      __builtin_prefetch(cH + o + (size_t)D_INNER * D_STATE, 0, 1);
    }
    Hst[o] = H;
    H = cA[o] * H + cH[o];
  }
}

// ---------------------------------------------------------------- scan pass 3: replay seeded, emit y = sum_n h*C
__global__ __launch_bounds__(256)
void k_scan3(const float* __restrict__ xc, const float* __restrict__ dt,
             const float* __restrict__ dbl, const float* __restrict__ A_log,
             const float* __restrict__ Hst, float* __restrict__ ys) {
  __shared__ __align__(16) float sdt[CHUNK][16];
  __shared__ __align__(16) float sxc[CHUNK][16];
  __shared__ __align__(16) float sbc[CHUNK][32];
  int tid = threadIdx.x;
  int n  = tid & 15;
  int dl = tid >> 4;
  int b  = blockIdx.y >> 5;
  int d0 = (blockIdx.y & 31) * 16;
  int d  = d0 + dl;
  int c  = blockIdx.x;
  size_t t0 = (size_t)b * SEQ + (size_t)c * CHUNK;
  stage_chunk(xc, dt, dbl, t0, d0, tid, sdt, sxc, sbc);

  float An = -__expf(A_log[d * 16 + n]);
  size_t o = ((((size_t)b * NCHUNK + c) * D_INNER) + d) * D_STATE + n;
  float h = Hst[o];
  for (int i = 0; i < CHUNK; ++i) {
    float dtv = sdt[i][dl];
    float u   = sxc[i][dl];
    float Bv  = sbc[i][n];
    float Cv  = sbc[i][16 + n];
    h = __expf(dtv * An) * h + dtv * u * Bv;
    float p = h * Cv;
    p += __shfl_xor(p, 1, 16);
    p += __shfl_xor(p, 2, 16);
    p += __shfl_xor(p, 4, 16);
    p += __shfl_xor(p, 8, 16);
    if (n == 0) ys[(t0 + i) * D_INNER + d] = p;
  }
}

// ---------------------------------------------------------------- gate: y = (ys + D*xc) * silu(z) -> bf16
__global__ __launch_bounds__(256)
void k_gate(const float* __restrict__ ys, const float* __restrict__ xc,
            const float* __restrict__ xiz, const float* __restrict__ Dp,
            unsigned short* __restrict__ yb) {
  int idx = blockIdx.x * 256 + threadIdx.x;    // BL * 512
  int d = idx & (D_INNER - 1);
  int t = idx >> 9;
  float z = xiz[(size_t)t * 1024 + D_INNER + d];
  float g = z / (1.0f + __expf(-z));
  yb[idx] = f2bf((ys[idx] + Dp[d] * xc[idx]) * g);
}

// ---------------------------------------------------------------- residual + (un)flip + channel-concat write
__global__ __launch_bounds__(256)
void k_final(const float* __restrict__ yo, const float* __restrict__ x,
             float* __restrict__ out, int flip, int dirOff) {
  int idx = blockIdx.x * 256 + threadIdx.x;    // BL * 256
  int c = idx & (D_MODEL - 1);
  int t = idx >> 8;
  int b = t / SEQ, tl = t - b * SEQ;
  int dtl = flip ? (SEQ - 1 - tl) : tl;
  size_t drow = (size_t)b * SEQ + dtl;
  out[drow * (2 * D_MODEL) + dirOff + c] =
      yo[(size_t)t * D_MODEL + c] + x[drow * D_MODEL + c];
}

// ================================================================ host
extern "C" void kernel_launch(void* const* d_in, const int* in_sizes, int n_in,
                              void* d_out, int out_size, void* d_ws, size_t ws_size,
                              hipStream_t stream) {
  (void)in_sizes; (void)n_in; (void)out_size; (void)ws_size;
  const float* x = (const float*)d_in[0];
  float* out = (float*)d_out;

  size_t off = 0;
  auto alloc = [&](size_t bytes) -> void* {
    size_t o = (off + 255) & ~(size_t)255;
    off = o + bytes;
    return (void*)((char*)d_ws + o);
  };

  for (int dir = 0; dir < 2; ++dir) {
    int p0 = 1 + dir * 10;
    const float* norm_w   = (const float*)d_in[p0 + 0];
    const float* in_proj  = (const float*)d_in[p0 + 1];
    const float* conv_w   = (const float*)d_in[p0 + 2];
    const float* conv_b   = (const float*)d_in[p0 + 3];
    const float* x_proj   = (const float*)d_in[p0 + 4];
    const float* dt_w     = (const float*)d_in[p0 + 5];
    const float* dt_b     = (const float*)d_in[p0 + 6];
    const float* A_log    = (const float*)d_in[p0 + 7];
    const float* Dp       = (const float*)d_in[p0 + 8];
    const float* out_proj = (const float*)d_in[p0 + 9];

    unsigned short* xn   = (unsigned short*)alloc((size_t)BL * D_MODEL * 2);
    unsigned short* win  = (unsigned short*)alloc((size_t)1024 * D_MODEL * 2);
    unsigned short* wbig = (unsigned short*)alloc((size_t)NBIG * D_INNER * 2);
    unsigned short* wout = (unsigned short*)alloc((size_t)D_MODEL * D_INNER * 2);
    float* xiz  = (float*)alloc((size_t)BL * 1024 * 4);
    float* xc   = (float*)alloc((size_t)BL * D_INNER * 4);
    unsigned short* xcb = (unsigned short*)alloc((size_t)BL * D_INNER * 2);
    float* dbl  = (float*)alloc((size_t)BL * NBIG * 4);
    float* dtv  = (float*)alloc((size_t)BL * D_INNER * 4);
    size_t csz  = (size_t)BATCH * NCHUNK * D_INNER * D_STATE * 4;
    float* cA   = (float*)alloc(csz);
    float* cH   = (float*)alloc(csz);
    float* Hst  = (float*)alloc(csz);
    float* ys   = (float*)alloc((size_t)BL * D_INNER * 4);
    unsigned short* yb = (unsigned short*)alloc((size_t)BL * D_INNER * 2);
    float* yo   = (float*)alloc((size_t)BL * D_MODEL * 4);

    k_rmsnorm<<<BL, 256, 0, stream>>>(x, norm_w, xn, dir);
    k_f2bf<<<(1024 * D_MODEL + 255) / 256, 256, 0, stream>>>(in_proj, win, 1024 * D_MODEL);
    k_wbig<<<(NBIG * D_INNER + 255) / 256, 256, 0, stream>>>(x_proj, dt_w, wbig);
    k_f2bf<<<(D_MODEL * D_INNER + 255) / 256, 256, 0, stream>>>(out_proj, wout, D_MODEL * D_INNER);

    // in_proj GEMM: [8192,1024] = xn[8192,256] * in_proj^T   (block tile 64M x 64N)
    k_gemm<<<dim3(BL / 64, 1024 / 64), 256, 0, stream>>>(xn, D_MODEL, win, D_MODEL,
                                                         xiz, 1024, D_MODEL, 4);
    k_conv<<<BL * D_INNER / 256, 256, 0, stream>>>(xiz, conv_w, conv_b, xc, xcb);

    // fused x_proj + dt projection: [8192,576] = xc[8192,512] * wbig^T
    k_gemm<<<dim3(BL / 64, NBIG / 64), 256, 0, stream>>>(xcb, D_INNER, wbig, D_INNER,
                                                         dbl, NBIG, D_INNER, 4);
    k_dt<<<BL * D_INNER / 256, 256, 0, stream>>>(dbl, dt_b, dtv);

    k_scan1<<<dim3(NCHUNK, BATCH * 32), 256, 0, stream>>>(xc, dtv, dbl, A_log, cA, cH);
    k_scan2<<<BATCH * 32, 256, 0, stream>>>(cA, cH, Hst);
    k_scan3<<<dim3(NCHUNK, BATCH * 32), 256, 0, stream>>>(xc, dtv, dbl, A_log, Hst, ys);

    k_gate<<<BL * D_INNER / 256, 256, 0, stream>>>(ys, xc, xiz, Dp, yb);

    // out_proj GEMM: [8192,256] = y[8192,512] * out_proj^T
    k_gemm<<<dim3(BL / 64, D_MODEL / 64), 256, 0, stream>>>(yb, D_INNER, wout, D_INNER,
                                                            yo, D_MODEL, D_INNER, 4);
    k_final<<<BL * D_MODEL / 256, 256, 0, stream>>>(yo, x, out, dir, dir * D_MODEL);
  }
}
